// GLOM_31808527794234
// MI455X (gfx1250) — matrix operand order repaired and checked by reading, not compiled, not run on hardware
//
#include <hip/hip_runtime.h>
#include <hip/hip_bf16.h>

// ---------------------------------------------------------------------------
// GLOM level-1 spatial-prior attention, fused flash-style for gfx1250 (MI455X)
//   B=2, C=128, H=W=64 (HW=4096), probs: 4096x4096 f32 (64MB, L2-resident)
//   out[b,c,i] = sum_j e_ij*p_ij*x[b,c,j] / (sum_j e_ij*p_ij + 1e-8*sum_j e_ij)
//   with e_ij = exp(cos_sim(i,j) - 1)   (cos_sim <= 1 -> fixed shift, 1 pass)
// Pipeline: double-buffered GLOBAL_LOAD_ASYNC_TO_LDS staging overlapped with
// v_wmma_f32_16x16x32_f16 QK^T and PV GEMMs.
// ---------------------------------------------------------------------------

typedef _Float16 half8  __attribute__((ext_vector_type(8)));
typedef _Float16 half16 __attribute__((ext_vector_type(16)));
typedef float    float8 __attribute__((ext_vector_type(8)));
typedef int      i32x4  __attribute__((ext_vector_type(4)));

#define HW   4096
#define CCH  128
#define NB   2
#define LOG2E 1.44269504088896f

#if defined(__has_builtin)
#if __has_builtin(__builtin_amdgcn_global_load_async_to_lds_b128) && \
    __has_builtin(__builtin_amdgcn_s_wait_asynccnt)
#define USE_ASYNC_LDS 1
#endif
#endif

#if defined(__has_builtin)
#if __has_builtin(__builtin_amdgcn_exp2f)
#define EXP2(x) __builtin_amdgcn_exp2f(x)   // raw v_exp_f32; arg in [-2.9, 0]
#else
#define EXP2(x) exp2f(x)
#endif
#else
#define EXP2(x) exp2f(x)
#endif

__device__ __forceinline__ half16 frag_from2(const _Float16* p0, const _Float16* p1) {
    half8 lo = *(const half8*)p0;
    half8 hi = *(const half8*)p1;
    return __builtin_shufflevector(lo, hi, 0,1,2,3,4,5,6,7,8,9,10,11,12,13,14,15);
}

// ---------------------------------------------------------------------------
// Prep 1: channel-normalize, write f16 xn[b][j][c] (position-major)
// ---------------------------------------------------------------------------
__global__ __launch_bounds__(128) void glom_norm_kernel(
    const float* __restrict__ embds, _Float16* __restrict__ xn16)
{
    const int j = blockIdx.x;          // spatial position 0..4095
    const int b = blockIdx.y;          // batch
    const int c = threadIdx.x;         // channel 0..127

    const float v = embds[((size_t)(b * CCH + c) << 12) + j];

    float ss = v * v;
    #pragma unroll
    for (int m = 16; m >= 1; m >>= 1) ss += __shfl_xor(ss, m);
    __shared__ float red[4];
    if ((threadIdx.x & 31) == 0) red[threadIdx.x >> 5] = ss;
    __syncthreads();
    const float ssq = red[0] + red[1] + red[2] + red[3];

    const float nrm = fmaxf(sqrtf(ssq), 1e-12f);
    xn16[((size_t)(b * HW + j) * CCH) + c] = (_Float16)(v / nrm);
}

// ---------------------------------------------------------------------------
// Prep 2: plain f16 cast of embds -> xvT[b][c][j] (identical layout to embds)
// ---------------------------------------------------------------------------
__global__ __launch_bounds__(256) void glom_cast_kernel(
    const float* __restrict__ embds, _Float16* __restrict__ xvT16, int n)
{
    int i = blockIdx.x * 256 + threadIdx.x;
    if (i < n) xvT16[i] = (_Float16)embds[i];
}

// ---------------------------------------------------------------------------
// Main fused attention kernel.
//   grid = (HW/128, B), block = 256 threads = 8 waves (wave32)
//   wave w owns query rows [I0 + 16w, I0 + 16w + 16)
// ---------------------------------------------------------------------------
__global__ __launch_bounds__(256) void glom_attn_kernel(
    const float*    __restrict__ probs,   // (4096, 4096)
    const _Float16* __restrict__ xn16,    // (B, 4096, 128) normalized
    const _Float16* __restrict__ xvT16,   // (B, 128, 4096) raw f16
    float*          __restrict__ out)     // (B, 128, 4096)
{
    const int tid  = threadIdx.x;
    const int wv   = tid >> 5;
    const int lane = tid & 31;
    const int g    = lane >> 4;       // half-wave group
    const int n    = lane & 15;       // column within tile (also A-frag row)
    const int b    = blockIdx.y;
    const int I0   = blockIdx.x * 128;

    const int irow0  = I0 + wv * 16 + g * 8;   // probs/out row for acc slot r
    const int kbaseA = g ? 16 : 0;             // A-fragment K base (contiguous layout)

    // Double-buffered LDS staging + per-wave P tile (72 KB of 320 KB/WGP)
    __shared__ __align__(16) _Float16 xn_s[2][64][CCH];   // 2 x 16 KB
    __shared__ __align__(16) _Float16 xv_s[2][CCH][64];   // 2 x 16 KB
    __shared__ __align__(16) _Float16 p_s[8][16][32];     //     8 KB

    // ---- tile staging: 8 x b128 per thread per buffer -------------------
    auto stage_tiles = [&](int buf, int J0) {
        // keys: 64 rows x 128 f16 contiguous = 1024 x b128
        const i32x4* gk = (const i32x4*)(xn16 + ((size_t)(b * HW + J0)) * CCH);
        i32x4*       lk = (i32x4*)&xn_s[buf][0][0];
        // values: channel-major rows of xvT, 128 x (64 f16 = 8 x b128)
        const int c = tid >> 1, h = tid & 1;
        const i32x4* gv = (const i32x4*)(xvT16 + ((size_t)(b * CCH + c) << 12) + J0 + h * 32);
        i32x4*       lv = (i32x4*)&xv_s[buf][c][h * 32];
#if defined(USE_ASYNC_LDS)
        #pragma unroll
        for (int t = 0; t < 4; ++t)
            __builtin_amdgcn_global_load_async_to_lds_b128(
                (i32x4*)(gk + tid + 256 * t), (i32x4*)(lk + tid + 256 * t), 0, 0);
        #pragma unroll
        for (int t = 0; t < 4; ++t)
            __builtin_amdgcn_global_load_async_to_lds_b128(
                (i32x4*)(gv + t), (i32x4*)(lv + t), 0, 0);
#else
        #pragma unroll
        for (int t = 0; t < 4; ++t) lk[tid + 256 * t] = gk[tid + 256 * t];
        #pragma unroll
        for (int t = 0; t < 4; ++t) lv[t] = gv[t];
#endif
    };

    // ---- load this wave's query A-fragments once (16 rows x 128 K) ------
    half16 afrag[4];
    {
        const _Float16* ap = xn16 + ((size_t)(b * HW + I0 + wv * 16 + n)) * CCH + kbaseA;
        #pragma unroll
        for (int q = 0; q < 4; ++q)
            afrag[q] = frag_from2(ap + q * 32, ap + q * 32 + 8);
    }

    float8 acc[8] = {};          // PV accumulators: row irow0+r, col 16*ct+n
    float  zacc[8] = {};         // partial sum exp(s-1)
    float  sacc[8] = {};         // partial sum exp(s-1)*probs

    stage_tiles(0, 0);           // prologue fill of buffer 0

    for (int J0 = 0; J0 < HW; J0 += 64) {
        const int cur = (J0 >> 6) & 1;
        const bool more = (J0 + 64) < HW;
        if (more) stage_tiles(cur ^ 1, J0 + 64);   // prefetch next block
#if defined(USE_ASYNC_LDS)
        if (more) __builtin_amdgcn_s_wait_asynccnt(8);  // cur buffer complete
        else      __builtin_amdgcn_s_wait_asynccnt(0);
#endif
        __syncthreads();

        #pragma unroll
        for (int jt = 0; jt < 2; ++jt) {
            // ---- QK^T: two 16x16 sims tiles (32 keys) -----------------
            #pragma unroll
            for (int sub = 0; sub < 2; ++sub) {
                const int jb = jt * 32 + sub * 16;
                float8 d = {};
                #pragma unroll
                for (int q = 0; q < 4; ++q) {
                    const _Float16* bp = &xn_s[cur][jb + n][q * 32 + g * 16];
                    half16 bfrag = *(const half16*)bp;
                    d = __builtin_amdgcn_wmma_f32_16x16x32_f16(
                            false, afrag[q], false, bfrag, (short)0, d, false, false);
                }
                // ---- exp(s-1), prior weight, row-sum partials ---------
                const int jcol = J0 + jb + n;
                const float* prow = probs + (size_t)irow0 * HW + jcol;
                #pragma unroll
                for (int r = 0; r < 8; ++r) {
                    float e = EXP2((d[r] - 1.0f) * LOG2E);
                    float w = e * prow[(size_t)r * HW];
                    zacc[r] += e;
                    sacc[r] += w;
                    p_s[wv][g * 8 + r][sub * 16 + n] = (_Float16)w;
                }
            }
            __syncthreads();   // P tile D-layout -> A-layout via LDS

            // ---- PV: P(16x32) x V(32x16) over 8 channel tiles ---------
            const _Float16* pp = &p_s[wv][n][kbaseA];
            half16 pfrag = frag_from2(pp, pp + 8);
            #pragma unroll
            for (int ct = 0; ct < 8; ++ct) {
                const _Float16* bp = &xv_s[cur][ct * 16 + n][jt * 32 + g * 16];
                half16 bfrag = *(const half16*)bp;
                acc[ct] = __builtin_amdgcn_wmma_f32_16x16x32_f16(
                              false, pfrag, false, bfrag, (short)0, acc[ct], false, false);
            }
            __syncthreads();   // protect p_s reuse next jt
        }
        __syncthreads();       // all reads of buffer `cur` done before refill
    }

    // ---- reduce row sums across the 16 lanes of each half-wave ----------
    float rden[8];
    #pragma unroll
    for (int r = 0; r < 8; ++r) {
        float z = zacc[r], s = sacc[r];
        #pragma unroll
        for (int m = 8; m >= 1; m >>= 1) { z += __shfl_xor(z, m); s += __shfl_xor(s, m); }
        rden[r] = 1.0f / (s + 1e-8f * z);
    }

    // ---- store out[b, c, irow0..irow0+7], c = 16*ct + n -----------------
    #pragma unroll
    for (int ct = 0; ct < 8; ++ct) {
        const int c = ct * 16 + n;
        float* op = out + ((size_t)(b * CCH + c) << 12) + irow0;
        float4 v0 = make_float4(acc[ct][0] * rden[0], acc[ct][1] * rden[1],
                                acc[ct][2] * rden[2], acc[ct][3] * rden[3]);
        float4 v1 = make_float4(acc[ct][4] * rden[4], acc[ct][5] * rden[5],
                                acc[ct][6] * rden[6], acc[ct][7] * rden[7]);
        *(float4*)(op)     = v0;
        *(float4*)(op + 4) = v1;
    }
}

// ---------------------------------------------------------------------------
extern "C" void kernel_launch(void* const* d_in, const int* in_sizes, int n_in,
                              void* d_out, int out_size, void* d_ws, size_t ws_size,
                              hipStream_t stream) {
    const float* embds = (const float*)d_in[0];   // (2,128,64,64) f32
    const float* probs = (const float*)d_in[1];   // (64,64,64,64) f32

    _Float16* xn16  = (_Float16*)d_ws;                        // 2 MB
    _Float16* xvT16 = xn16 + (size_t)NB * HW * CCH;           // 2 MB

    glom_norm_kernel<<<dim3(HW, NB), 128, 0, stream>>>(embds, xn16);
    const int nelem = NB * CCH * HW;
    glom_cast_kernel<<<(nelem + 255) / 256, 256, 0, stream>>>(embds, xvT16, nelem);

    glom_attn_kernel<<<dim3(HW / 128, NB), 256, 0, stream>>>(
        probs, xn16, xvT16, (float*)d_out);
}
